// WindowAttention_17145509445839
// MI455X (gfx1250) — compile-verified, matrix operand order checked by reading, not run on hardware
//
#include <hip/hip_runtime.h>
#include <hip/hip_bf16.h>

typedef __attribute__((ext_vector_type(16))) _Float16 h16;
typedef __attribute__((ext_vector_type(8)))  _Float16 h8;
typedef __attribute__((ext_vector_type(8)))  float    f8;
typedef __attribute__((ext_vector_type(4)))  unsigned int u32x4;
typedef __attribute__((ext_vector_type(8)))  int      i32x8;
typedef __attribute__((ext_vector_type(4)))  int      i32x4;

#define NTOK   49
#define CH     512
#define NH     16
#define HDIM   32
#define NWIN   64
#define M_ROWS (4096*49)             /* 200704, divisible by 128 */
#define SCALE_Q 0.17677669529663687f /* 32^-0.5 */

#if defined(__has_builtin)
#if __has_builtin(__builtin_amdgcn_tensor_load_to_lds) && __has_builtin(__builtin_amdgcn_s_wait_tensorcnt)
#define HAVE_TDM 1
#endif
#endif
#ifndef HAVE_TDM
#define HAVE_TDM 0
#endif

__device__ __forceinline__ h16 cat16(h8 lo, h8 hi) {
    return __builtin_shufflevector(lo, hi, 0,1,2,3,4,5,6,7,8,9,10,11,12,13,14,15);
}

__device__ __forceinline__ h16 frag_pred(const _Float16* __restrict__ p, int kbase, bool valid) {
    h8 z = {};
    h8 lo = valid ? *(const h8*)(p + kbase)      : z;
    h8 hi = valid ? *(const h8*)(p + kbase + 16) : z;
    return cat16(lo, hi);
}

__device__ __forceinline__ h8 cvt8(float4 a, float4 b) {
    h8 r;
    r[0] = (_Float16)a.x; r[1] = (_Float16)a.y; r[2] = (_Float16)a.z; r[3] = (_Float16)a.w;
    r[4] = (_Float16)b.x; r[5] = (_Float16)b.y; r[6] = (_Float16)b.z; r[7] = (_Float16)b.w;
    return r;
}

#if HAVE_TDM
// TDM 2D tile load: global (row-major, row_stride_dw dwords/row) -> LDS, data_size=4B.
__device__ __forceinline__ void tdm_load_tile_2d(unsigned lds_off, const void* gaddr,
                                                 unsigned tile_dw_x, unsigned tile_rows,
                                                 unsigned row_stride_dw)
{
    unsigned long long ga = (unsigned long long)(uintptr_t)gaddr;
    const unsigned td1 = 0x100000u;                  // large dim1 extent
    u32x4 g0;
    g0[0] = 1u;                                      // count=1, user descriptor
    g0[1] = lds_off;                                 // lds_addr (bytes)
    g0[2] = (unsigned)ga;                            // global_addr[31:0]
    g0[3] = (unsigned)((ga >> 32) & 0x1ffffffu) | (2u << 30); // addr[56:32] | type=2
    i32x8 g1;
    g1[0] = (int)(2u << 16);                                     // data_size code 2 = 4B
    g1[1] = (int)((tile_dw_x & 0xffffu) << 16);                  // tensor_dim0 lo16 @bit48
    g1[2] = (int)((tile_dw_x >> 16) | ((td1 & 0xffffu) << 16));  // td0 hi | td1 lo
    g1[3] = (int)((td1 >> 16) | ((tile_dw_x & 0xffffu) << 16));  // td1 hi | tile_dim0 @bit112
    g1[4] = (int)(tile_rows & 0xffffu);                          // tile_dim1 | tile_dim2=0
    g1[5] = (int)row_stride_dw;                                  // tensor_dim0_stride lo32
    g1[6] = 0;
    g1[7] = 0;
    i32x4 gz = {0, 0, 0, 0};
#if defined(__clang_major__) && (__clang_major__ >= 23)
    i32x8 gz8 = {0, 0, 0, 0, 0, 0, 0, 0};
    __builtin_amdgcn_tensor_load_to_lds(g0, g1, gz, gz, gz8, 0);
#else
    __builtin_amdgcn_tensor_load_to_lds(g0, g1, gz, gz, 0);
#endif
}
#endif

// ---------------- prep: weight transpose->f16, bias gather ----------------
__global__ __launch_bounds__(256) void k_prep(
    const float* __restrict__ w_qkv, const float* __restrict__ w_proj,
    const float* __restrict__ tab,   const int* __restrict__ idx,
    _Float16* __restrict__ wqkvT, _Float16* __restrict__ wprojT,
    float* __restrict__ biasF)
{
    int i = blockIdx.x * 256 + threadIdx.x;
    if (i < 512*1536) {
        int k = i / 1536, c = i % 1536;
        wqkvT[(size_t)c*512 + k] = (_Float16)w_qkv[i];
    }
    int j = i - 512*1536;
    if (j >= 0 && j < 512*512) {
        int k = j >> 9, c = j & 511;
        wprojT[(size_t)c*512 + k] = (_Float16)w_proj[j];
    }
    int m = i - 512*1536 - 512*512;
    if (m >= 0 && m < NH*NTOK*NTOK) {
        int h = m / (NTOK*NTOK), r = m % (NTOK*NTOK);
        biasF[m] = tab[idx[r]*NH + h];
    }
}

// ---------------- QKV GEMM: 128x64 block tile, K chunks of 64, LDS double buffer ----------------
__global__ __launch_bounds__(256) void k_qkv_gemm(
    const float* __restrict__ x, const _Float16* __restrict__ wT,
    const float* __restrict__ bqkv,
    _Float16* __restrict__ q, _Float16* __restrict__ kmat, _Float16* __restrict__ v)
{
    __shared__ __align__(16) _Float16 As[2][128*64];  // 2 x 16 KB
    __shared__ __align__(16) _Float16 Bs[2][64*64];   // 2 x 8 KB
    const int t = threadIdx.x;
    const int wave = t >> 5, lane = t & 31;
    const int hl = lane >> 4, ln = lane & 15, kbase = hl*8;
    const int nb0  = blockIdx.x * 64;    // output column block
    const int row0 = blockIdx.y * 128;   // output row block

    // A staging: thread -> row t>>1, halves [(t&1)*32 .. +31] of the 64-wide K chunk
    const int arow = t >> 1, aoff = (t & 1) * 32;
    const float* agbase = x + (size_t)(row0 + arow)*512 + aoff;

    // ---- stage chunk 0 ----
    {
        const float4* ag = (const float4*)agbase;
        float4 ar[8];
        #pragma unroll
        for (int i = 0; i < 8; ++i) ar[i] = ag[i];
#if HAVE_TDM
        if (wave == 0)
            tdm_load_tile_2d((unsigned)(uintptr_t)&Bs[0][0], wT + (size_t)nb0*512, 32, 64, 256);
#else
        {
            int bcol = t >> 2, boff = (t & 3) * 16;
            const h8* bg = (const h8*)(wT + (size_t)(nb0 + bcol)*512 + boff);
            h8 b0 = bg[0], b1 = bg[1];
            *(h8*)&Bs[0][bcol*64 + boff]     = b0;
            *(h8*)&Bs[0][bcol*64 + boff + 8] = b1;
        }
#endif
        #pragma unroll
        for (int j = 0; j < 4; ++j)
            *(h8*)&As[0][arow*64 + aoff + j*8] = cvt8(ar[2*j], ar[2*j+1]);
#if HAVE_TDM
        if (wave == 0) __builtin_amdgcn_s_wait_tensorcnt(0);
#endif
    }
    __syncthreads();

    f8 fz = {};
    f8 acc[4] = {fz, fz, fz, fz};
    for (int kc = 0; kc < 8; ++kc) {
        const int cur = kc & 1, nxt = cur ^ 1;
        float4 ar[8];
#if !HAVE_TDM
        h8 br0, br1;
        const int bcol = t >> 2, boff = (t & 3) * 16;
#endif
        if (kc < 7) {   // prefetch chunk kc+1 (overlaps the compute below)
            const float4* ag = (const float4*)(agbase + (kc + 1)*64);
            #pragma unroll
            for (int i = 0; i < 8; ++i) ar[i] = ag[i];
#if HAVE_TDM
            if (wave == 0)
                tdm_load_tile_2d((unsigned)(uintptr_t)&Bs[nxt][0],
                                 wT + (size_t)nb0*512 + (kc + 1)*64, 32, 64, 256);
#else
            {
                const h8* bg = (const h8*)(wT + (size_t)(nb0 + bcol)*512 + (kc + 1)*64 + boff);
                br0 = bg[0]; br1 = bg[1];
            }
#endif
        }
        // ---- compute on current buffers: 2 K-steps x 4 col tiles ----
        const _Float16* ab = &As[cur][(wave*16 + ln)*64 + kbase];
        const _Float16* bb = &Bs[cur][ln*64 + kbase];
        #pragma unroll
        for (int ks = 0; ks < 2; ++ks) {
            h16 a = cat16(*(const h8*)(ab + ks*32), *(const h8*)(ab + ks*32 + 16));
            #pragma unroll
            for (int bi = 0; bi < 4; ++bi) {
                const _Float16* bp = bb + bi*16*64 + ks*32;
                h16 bfr = cat16(*(const h8*)bp, *(const h8*)(bp + 16));
                acc[bi] = __builtin_amdgcn_wmma_f32_16x16x32_f16(
                    false, a, false, bfr, (short)0, acc[bi], false, false);
            }
        }
        __syncthreads();
        if (kc < 7) {   // publish prefetched chunk
            #pragma unroll
            for (int j = 0; j < 4; ++j)
                *(h8*)&As[nxt][arow*64 + aoff + j*8] = cvt8(ar[2*j], ar[2*j+1]);
#if HAVE_TDM
            if (wave == 0) __builtin_amdgcn_s_wait_tensorcnt(0);
#else
            *(h8*)&Bs[nxt][bcol*64 + boff]     = br0;
            *(h8*)&Bs[nxt][bcol*64 + boff + 8] = br1;
#endif
            __syncthreads();
        }
    }

    // ---- epilogue: bias, q-scale, scatter into [B][H][N][HD] f16 ----
    #pragma unroll
    for (int bi = 0; bi < 4; ++bi) {
        int col  = nb0 + bi*16 + ln;
        int sel  = col >> 9;                    // 0=q 1=k 2=v
        int cc   = col & 511;
        int head = cc >> 5, hd = cc & 31;
        float bv = bqkv[col];
        _Float16* dst = (sel == 0) ? q : (sel == 1) ? kmat : v;
        float sc = (sel == 0) ? SCALE_Q : 1.0f;
        #pragma unroll
        for (int r = 0; r < 8; ++r) {
            int grow = row0 + wave*16 + 8*hl + r;
            int b_ = grow / NTOK, n_ = grow - b_*NTOK;
            float val = (acc[bi][r] + bv) * sc;
            dst[(((size_t)b_*NH + head)*NTOK + n_)*HDIM + hd] = (_Float16)val;
        }
    }
}

// ---------------- fused attention: scores + bias + mask + softmax + p@v ----------------
__global__ __launch_bounds__(128) void k_attn(
    const _Float16* __restrict__ qs, const _Float16* __restrict__ ks,
    const _Float16* __restrict__ vs, const float* __restrict__ biasF,
    const float* __restrict__ mask,  _Float16* __restrict__ out)
{
    __shared__ __align__(16) _Float16 lds[4*(32*64 + 64*64)];
    const int t = threadIdx.x;
    const int wave = t >> 5, lane = t & 31;
    const int hl = lane >> 4, ln = lane & 15, kbase = hl*8;
    const int b = blockIdx.x;
    const int w = b & (NWIN - 1);
    _Float16* vT = lds + wave*(32*64 + 64*64);
    _Float16* pr = vT + 32*64;
    f8 fz = {};

    for (int h = wave; h < NH; h += 4) {
        const size_t base = ((size_t)b*NH + h) * NTOK * HDIM;
        const _Float16* qh = qs + base;
        const _Float16* kh = ks + base;
        const _Float16* vh = vs + base;

        for (int i = lane; i < 64*32; i += 32) {
            int tok = i >> 5, hd = i & 31;
            vT[hd*64 + tok] = (tok < NTOK) ? vh[i] : (_Float16)0.0f;
        }

        f8 S[4][4];
        #pragma unroll
        for (int mi = 0; mi < 4; ++mi) {
            int qrow = mi*16 + ln;
            h16 a = frag_pred(qh + qrow*HDIM, kbase, qrow < NTOK);
            #pragma unroll
            for (int ni = 0; ni < 4; ++ni) {
                int krow = ni*16 + ln;
                h16 bf = frag_pred(kh + krow*HDIM, kbase, krow < NTOK);
                S[mi][ni] = __builtin_amdgcn_wmma_f32_16x16x32_f16(
                    false, a, false, bf, (short)0, fz, false, false);
            }
        }

        const float* bh = biasF + h*NTOK*NTOK;
        const float* mw = mask  + w*NTOK*NTOK;
        #pragma unroll
        for (int mi = 0; mi < 4; ++mi) {
            #pragma unroll
            for (int r = 0; r < 8; ++r) {
                int row = mi*16 + 8*hl + r;
                float sv[4];
                float mx = -1e30f;
                #pragma unroll
                for (int ni = 0; ni < 4; ++ni) {
                    int col = ni*16 + ln;
                    float val = S[mi][ni][r];
                    if (row < NTOK && col < NTOK)
                        val += bh[row*NTOK + col] + mw[row*NTOK + col];
                    else
                        val = -1e30f;
                    sv[ni] = val;
                    mx = fmaxf(mx, val);
                }
                for (int d = 1; d < 16; d <<= 1) mx = fmaxf(mx, __shfl_xor(mx, d, 32));
                float sum = 0.f;
                #pragma unroll
                for (int ni = 0; ni < 4; ++ni) { float e = __expf(sv[ni] - mx); sv[ni] = e; sum += e; }
                for (int d = 1; d < 16; d <<= 1) sum += __shfl_xor(sum, d, 32);
                float inv = 1.0f / sum;
                #pragma unroll
                for (int ni = 0; ni < 4; ++ni)
                    pr[row*64 + ni*16 + ln] = (_Float16)(sv[ni] * inv);
            }
        }

        #pragma unroll
        for (int mi = 0; mi < 4; ++mi) {
            f8 O0 = fz, O1 = fz;
            #pragma unroll
            for (int kc = 0; kc < 2; ++kc) {
                const _Float16* pp = pr + (mi*16 + ln)*64 + kc*32 + kbase;
                h16 a = cat16(*(const h8*)pp, *(const h8*)(pp + 16));
                const _Float16* v0 = vT + ln*64        + kc*32 + kbase;
                const _Float16* v1 = vT + (16 + ln)*64 + kc*32 + kbase;
                h16 b0 = cat16(*(const h8*)v0, *(const h8*)(v0 + 16));
                h16 b1 = cat16(*(const h8*)v1, *(const h8*)(v1 + 16));
                O0 = __builtin_amdgcn_wmma_f32_16x16x32_f16(false, a, false, b0, (short)0, O0, false, false);
                O1 = __builtin_amdgcn_wmma_f32_16x16x32_f16(false, a, false, b1, (short)0, O1, false, false);
            }
            #pragma unroll
            for (int r = 0; r < 8; ++r) {
                int tok = mi*16 + 8*hl + r;
                if (tok < NTOK) {
                    size_t ob = ((size_t)b*NTOK + tok)*CH + h*HDIM;
                    out[ob + ln]      = (_Float16)O0[r];
                    out[ob + 16 + ln] = (_Float16)O1[r];
                }
            }
        }
    }
}

// ---------------- output projection: same pipelined skeleton, TDM for A and B ----------------
__global__ __launch_bounds__(256) void k_proj_gemm(
    const _Float16* __restrict__ A, const _Float16* __restrict__ wT,
    const float* __restrict__ bp, float* __restrict__ out)
{
    __shared__ __align__(16) _Float16 As[2][128*64];
    __shared__ __align__(16) _Float16 Bs[2][64*64];
    const int t = threadIdx.x;
    const int wave = t >> 5, lane = t & 31;
    const int hl = lane >> 4, ln = lane & 15, kbase = hl*8;
    const int nb0  = blockIdx.x * 64;
    const int row0 = blockIdx.y * 128;

    const int arow = t >> 1, aoff = (t & 1) * 32;   // manual-staging assignment
    const _Float16* agbase = A + (size_t)(row0 + arow)*512 + aoff;

    // ---- stage chunk 0 ----
#if HAVE_TDM
    if (wave == 0) {
        tdm_load_tile_2d((unsigned)(uintptr_t)&As[0][0], A + (size_t)row0*512, 32, 128, 256);
        tdm_load_tile_2d((unsigned)(uintptr_t)&Bs[0][0], wT + (size_t)nb0*512, 32, 64, 256);
        __builtin_amdgcn_s_wait_tensorcnt(0);
    }
#else
    {
        const h8* ag = (const h8*)agbase;
        h8 a0 = ag[0], a1 = ag[1], a2 = ag[2], a3 = ag[3];
        int bcol = t >> 2, boff = (t & 3) * 16;
        const h8* bg = (const h8*)(wT + (size_t)(nb0 + bcol)*512 + boff);
        h8 b0 = bg[0], b1 = bg[1];
        h8* ad = (h8*)&As[0][arow*64 + aoff];
        ad[0] = a0; ad[1] = a1; ad[2] = a2; ad[3] = a3;
        *(h8*)&Bs[0][bcol*64 + boff]     = b0;
        *(h8*)&Bs[0][bcol*64 + boff + 8] = b1;
    }
#endif
    __syncthreads();

    f8 fz = {};
    f8 acc[4] = {fz, fz, fz, fz};
    for (int kc = 0; kc < 8; ++kc) {
        const int cur = kc & 1, nxt = cur ^ 1;
#if !HAVE_TDM
        h8 ar0, ar1, ar2, ar3, br0, br1;
        const int bcol = t >> 2, boff = (t & 3) * 16;
#endif
        if (kc < 7) {   // prefetch chunk kc+1
#if HAVE_TDM
            if (wave == 0) {
                tdm_load_tile_2d((unsigned)(uintptr_t)&As[nxt][0],
                                 A + (size_t)row0*512 + (kc + 1)*64, 32, 128, 256);
                tdm_load_tile_2d((unsigned)(uintptr_t)&Bs[nxt][0],
                                 wT + (size_t)nb0*512 + (kc + 1)*64, 32, 64, 256);
            }
#else
            {
                const h8* ag = (const h8*)(agbase + (kc + 1)*64);
                ar0 = ag[0]; ar1 = ag[1]; ar2 = ag[2]; ar3 = ag[3];
                const h8* bg = (const h8*)(wT + (size_t)(nb0 + bcol)*512 + (kc + 1)*64 + boff);
                br0 = bg[0]; br1 = bg[1];
            }
#endif
        }
        const _Float16* ab = &As[cur][(wave*16 + ln)*64 + kbase];
        const _Float16* bb = &Bs[cur][ln*64 + kbase];
        #pragma unroll
        for (int ks = 0; ks < 2; ++ks) {
            h16 a = cat16(*(const h8*)(ab + ks*32), *(const h8*)(ab + ks*32 + 16));
            #pragma unroll
            for (int bi = 0; bi < 4; ++bi) {
                const _Float16* bpp = bb + bi*16*64 + ks*32;
                h16 bfr = cat16(*(const h8*)bpp, *(const h8*)(bpp + 16));
                acc[bi] = __builtin_amdgcn_wmma_f32_16x16x32_f16(
                    false, a, false, bfr, (short)0, acc[bi], false, false);
            }
        }
        __syncthreads();
        if (kc < 7) {
#if HAVE_TDM
            if (wave == 0) __builtin_amdgcn_s_wait_tensorcnt(0);
#else
            h8* ad = (h8*)&As[nxt][arow*64 + aoff];
            ad[0] = ar0; ad[1] = ar1; ad[2] = ar2; ad[3] = ar3;
            *(h8*)&Bs[nxt][bcol*64 + boff]     = br0;
            *(h8*)&Bs[nxt][bcol*64 + boff + 8] = br1;
#endif
            __syncthreads();
        }
    }

    #pragma unroll
    for (int bi = 0; bi < 4; ++bi) {
        int col = nb0 + bi*16 + ln;
        float bv = bp[col];
        #pragma unroll
        for (int r = 0; r < 8; ++r) {
            int grow = row0 + wave*16 + 8*hl + r;
            out[(size_t)grow*512 + col] = acc[bi][r] + bv;
        }
    }
}

extern "C" void kernel_launch(void* const* d_in, const int* in_sizes, int n_in,
                              void* d_out, int out_size, void* d_ws, size_t ws_size,
                              hipStream_t stream) {
    (void)in_sizes; (void)n_in; (void)out_size; (void)ws_size;
    const float* x      = (const float*)d_in[0];
    const float* mask   = (const float*)d_in[1];
    const float* w_qkv  = (const float*)d_in[2];
    const float* b_qkv  = (const float*)d_in[3];
    const float* w_proj = (const float*)d_in[4];
    const float* b_proj = (const float*)d_in[5];
    const float* ptab   = (const float*)d_in[6];
    const int*   pidx   = (const int*)d_in[7];

    char* ws = (char*)d_ws;
    size_t off = 0;
    auto take = [&](size_t bytes) -> void* {
        void* p = ws + off;
        off += (bytes + 255) & ~(size_t)255;
        return p;
    };
    _Float16* wqkvT  = (_Float16*)take((size_t)1536*512*2);
    _Float16* wprojT = (_Float16*)take((size_t)512*512*2);
    float*    biasF  = (float*)   take((size_t)NH*NTOK*NTOK*4);
    const size_t qkvE = (size_t)4096*NH*NTOK*HDIM;
    _Float16* q  = (_Float16*)take(qkvE*2);
    _Float16* k  = (_Float16*)take(qkvE*2);
    _Float16* v  = (_Float16*)take(qkvE*2);
    _Float16* ao = (_Float16*)take((size_t)M_ROWS*CH*2);

    const int prepN = 512*1536 + 512*512 + NH*NTOK*NTOK;
    k_prep<<<(prepN + 255)/256, 256, 0, stream>>>(w_qkv, w_proj, ptab, pidx, wqkvT, wprojT, biasF);
    k_qkv_gemm<<<dim3(24, M_ROWS/128), 256, 0, stream>>>(x, wqkvT, b_qkv, q, k, v);
    k_attn<<<4096, 128, 0, stream>>>(q, k, v, biasF, mask, ao);
    k_proj_gemm<<<dim3(8, M_ROWS/128), 256, 0, stream>>>(ao, wprojT, b_proj, (float*)d_out);
}
